// SS2DTemWithoutDiff_39848706573422
// MI455X (gfx1250) — matrix operand order repaired and checked by reading, not compiled, not run on hardware
//
#include <hip/hip_runtime.h>
#include <hip/hip_bf16.h>

// ---------------------------------------------------------------------------
// SS2D forward for MI455X (gfx1250). GEMMs via v_wmma_f32_16x16x32_bf16 with
// branch-free clamped b128 tile loads + register double-buffering; selective
// scan in fp32 VALU with coalesced layouts.
// ---------------------------------------------------------------------------

#define B_ 4
#define H_ 48
#define W_ 48
#define DM 192
#define DI 384
#define DS_ 16
#define DTR 12
#define KD 4
#define L_ (H_ * W_)          // 2304
#define M_TOT (B_ * L_)       // 9216
#define C2 (DTR + 2 * DS_)    // 44

typedef __attribute__((ext_vector_type(16))) __bf16 v16bf;
typedef __attribute__((ext_vector_type(8)))  __bf16 v8bf;
typedef __attribute__((ext_vector_type(4)))  __bf16 v4bf;
typedef __attribute__((ext_vector_type(8)))  float  v8f;

// ------------------------------ WMMA GEMM ----------------------------------
// C[m,n] = sum_k A[m,k] * B(k,n).
//   BT=true : B given as Bt[n][k]  (weights)
//   BT=false: B given as B[k][n]   (k-major activations; transposed on LDS
//             store so fragment reads stay contiguous)
// Batched over blockIdx.z: A += (z % aMod)*aStride, B += z*bStride,
// C += z*cStride.
//
// Out-of-tile handling is branch-free and load-unconditional:
//  * all addresses are clamped with min() (legal: lda/ldb/K multiples of 4),
//  * m/n overruns produce garbage only in C rows/cols that are never stored,
//  * K overruns (only the !BT dts case, K=12) are killed by multiplying the
//    B rows with a per-row 0/1 mask (uniform across the vec4), so A's clamped
//    garbage columns hit zeroed B rows.
#define MT 64
#define NT 64
#define KT 32
#define LDP 40   // LDS row pad (80B = 20 banks -> conflict-free b128 reads)

template <bool BT>
__global__ __launch_bounds__(256) void wmma_gemm_kernel(
    const float* __restrict__ Ag, const float* __restrict__ Bg,
    float* __restrict__ Cg, int M, int N, int Kdim, int lda, int ldb, int ldc,
    long aStride, int aMod, long bStride, long cStride) {
  __shared__ __align__(16) __bf16 sA[MT][LDP];
  __shared__ __align__(16) __bf16 sB[NT][LDP];

  const int z = blockIdx.z;
  const float* A = Ag + (long)(z % aMod) * aStride;
  const float* B = Bg + (long)z * bStride;
  float* C = Cg + (long)z * cStride;

  const int n0 = blockIdx.x * NT;
  const int m0 = blockIdx.y * MT;
  const int t = threadIdx.x;
  const int lane = t & 31;
  const int wv = t >> 5;          // wave 0..7
  const int wm = (wv & 3) * 16;   // wave m-offset in tile
  const int wn = (wv >> 2) * 32;  // wave n-offset in tile
  const int hf = lane >> 4;       // half-wave select
  const int l16 = lane & 15;

  // vec4-granularity cooperative load coordinates
  const int ar = t >> 3;         // 0..31 row (A / Bt), second pass +32
  const int ac4 = (t & 7) * 4;   // k offset within chunk
  const int bn4 = (t & 15) * 4;  // !BT: n offset within tile
  const int bk = t >> 4;         // !BT: 0..15 k row, second pass +16

  const int kClamp = (Kdim >= 4) ? (Kdim - 4) : 0;
  const int nClamp = (N >= 4) ? (N - 4) : 0;

  v8f acc0 = {}, acc1 = {};
  float4 aR[2], bR[2];

  auto loadA = [&](int k0) {
#pragma unroll
    for (int p = 0; p < 2; ++p) {
      const int m = m0 + ar + p * 32;
      const int kk = k0 + ac4;
      const int mc = (m < M) ? m : (M - 1);       // garbage rows never stored
      const int kc = (kk <= kClamp) ? kk : kClamp; // garbage k hits zeroed B
      aR[p] = *(const float4*)(A + (long)mc * lda + kc);
    }
  };
  auto loadB = [&](int k0) {
#pragma unroll
    for (int p = 0; p < 2; ++p) {
      if (BT) {
        const int n = n0 + ar + p * 32;
        const int kk = k0 + ac4;
        const int nc = (n < N) ? n : (N - 1);
        const int kc = (kk <= kClamp) ? kk : kClamp;
        bR[p] = *(const float4*)(B + (long)nc * ldb + kc);
      } else {
        const int kRow = k0 + bk + p * 16;
        const int nn = n0 + bn4;
        const int kc = (kRow < Kdim) ? kRow : (Kdim - 1);
        const int nc = (nn <= nClamp) ? nn : nClamp;
        float4 v = *(const float4*)(B + (long)kc * ldb + nc);
        const float kmask = (kRow < Kdim) ? 1.f : 0.f;  // uniform per vec4
        v.x *= kmask;
        v.y *= kmask;
        v.z *= kmask;
        v.w *= kmask;
        bR[p] = v;
      }
    }
  };
  auto stage = [&]() {
#pragma unroll
    for (int p = 0; p < 2; ++p) {
      v4bf oa;
      oa[0] = (__bf16)aR[p].x;
      oa[1] = (__bf16)aR[p].y;
      oa[2] = (__bf16)aR[p].z;
      oa[3] = (__bf16)aR[p].w;
      *(v4bf*)&sA[ar + p * 32][ac4] = oa;
      if (BT) {
        v4bf ob;
        ob[0] = (__bf16)bR[p].x;
        ob[1] = (__bf16)bR[p].y;
        ob[2] = (__bf16)bR[p].z;
        ob[3] = (__bf16)bR[p].w;
        *(v4bf*)&sB[ar + p * 32][ac4] = ob;
      } else {
        const int kkk = bk + p * 16;  // transposed scatter -> sB[n][k]
        sB[bn4 + 0][kkk] = (__bf16)bR[p].x;
        sB[bn4 + 1][kkk] = (__bf16)bR[p].y;
        sB[bn4 + 2][kkk] = (__bf16)bR[p].z;
        sB[bn4 + 3][kkk] = (__bf16)bR[p].w;
      }
    }
  };

  const int nK = (Kdim + KT - 1) / KT;
  loadA(0);
  loadB(0);
  for (int kt = 0; kt < nK; ++kt) {
    stage();
    __syncthreads();
    if (kt + 1 < nK) {  // fetch next chunk while this chunk's WMMAs run
      loadA((kt + 1) * KT);
      loadB((kt + 1) * KT);
    }
    if (kt + 2 < nK) {  // speculative prefetch two chunks ahead (gfx1250)
      __builtin_prefetch(A + (long)(m0 + ar) * lda + (kt + 2) * KT + ac4, 0, 1);
      if (BT)
        __builtin_prefetch(B + (long)(n0 + ar) * ldb + (kt + 2) * KT + ac4, 0,
                           1);
      else
        __builtin_prefetch(B + (long)((kt + 2) * KT + bk) * ldb + n0 + bn4, 0,
                           1);
    }

    // ---- A fragment (16x32 bf16, contiguous k order, same as B):
    // lane<16: e0..15 -> K0..15 ;
    // lane>=16: e0..15 -> K16..31   (M = lane&15)
    const __bf16* arow = &sA[wm + l16][0];
    v8bf alo = *(const v8bf*)(arow + hf * 16);
    v8bf ahi = *(const v8bf*)(arow + hf * 16 + 8);
    v16bf av = __builtin_shufflevector(alo, ahi, 0, 1, 2, 3, 4, 5, 6, 7, 8, 9,
                                       10, 11, 12, 13, 14, 15);
    // ---- B fragments (32x16): lane<16 -> K0..15, lane>=16 -> K16..31,
    // N = lane&15; sB is B-transposed so reads are contiguous.
    const __bf16* brow0 = &sB[wn + l16][0];
    v8bf b0lo = *(const v8bf*)(brow0 + hf * 16);
    v8bf b0hi = *(const v8bf*)(brow0 + hf * 16 + 8);
    v16bf bv0 = __builtin_shufflevector(b0lo, b0hi, 0, 1, 2, 3, 4, 5, 6, 7, 8,
                                        9, 10, 11, 12, 13, 14, 15);
    const __bf16* brow1 = &sB[wn + 16 + l16][0];
    v8bf b1lo = *(const v8bf*)(brow1 + hf * 16);
    v8bf b1hi = *(const v8bf*)(brow1 + hf * 16 + 8);
    v16bf bv1 = __builtin_shufflevector(b1lo, b1hi, 0, 1, 2, 3, 4, 5, 6, 7, 8,
                                        9, 10, 11, 12, 13, 14, 15);

    acc0 = __builtin_amdgcn_wmma_f32_16x16x32_bf16(false, av, false, bv0,
                                                   (short)0, acc0, false,
                                                   false);
    acc1 = __builtin_amdgcn_wmma_f32_16x16x32_bf16(false, av, false, bv1,
                                                   (short)0, acc1, false,
                                                   false);
    __syncthreads();
  }

  // ---- C store: VGPR v -> M = v (lanes 0-15) / v+8 (lanes 16-31), N = lane&15
#pragma unroll
  for (int v = 0; v < 8; ++v) {
    const int m = m0 + wm + v + hf * 8;
    if (m < M) {
      int n = n0 + wn + l16;
      if (n < N) C[(long)m * ldc + n] = acc0[v];
      n += 16;
      if (n < N) C[(long)m * ldc + n] = acc1[v];
    }
  }
}

// --------------------------- depthwise/grouped conv ------------------------
// out[b,c,h,w] = silu( bias[c] + sum_{3x3} in[b,ih,iw, c>>chShift] * w[c,kh,kw] )
// in is channel-last with leading dim inLd. out is [B][DI][H][W].
__global__ void conv_dw_kernel(const float* __restrict__ in, int inLd,
                               int chShift, const float* __restrict__ wgt,
                               const float* __restrict__ bias,
                               float* __restrict__ out) {
  const int idx = blockIdx.x * 256 + threadIdx.x;
  if (idx >= B_ * H_ * W_ * DI) return;
  const int c = idx % DI;
  const int w = (idx / DI) % W_;
  const int h = (idx / (DI * W_)) % H_;
  const int b = idx / (DI * W_ * H_);
  const int ci = c >> chShift;
  float s = bias[c];
#pragma unroll
  for (int kh = 0; kh < 3; ++kh) {
    const int ih = h + kh - 1;
    if (ih < 0 || ih >= H_) continue;
#pragma unroll
    for (int kw = 0; kw < 3; ++kw) {
      const int iw = w + kw - 1;
      if (iw < 0 || iw >= W_) continue;
      s += in[((long)((b * H_ + ih) * W_ + iw)) * inLd + ci] *
           wgt[c * 9 + kh * 3 + kw];
    }
  }
  out[(((long)(b * DI + c) * H_) + h) * W_ + w] = s / (1.f + __expf(-s));
}

// ---------------------- directional interleave (xs, u_t) -------------------
// xs layout [b][k][c][l], ut layout [b][k][l][c]
__global__ void interp_kernel(const float* __restrict__ xc,
                              const float* __restrict__ xtc,
                              float* __restrict__ xs, float* __restrict__ ut) {
  const int idx = blockIdx.x * 256 + threadIdx.x;
  if (idx >= B_ * H_ * W_ * DI) return;
  const int c = idx % DI;
  const int w = (idx / DI) % W_;
  const int h = (idx / (DI * W_)) % H_;
  const int b = idx / (DI * W_ * H_);
  const long pc = (long)(b * DI + c) * L_;

  // inter_col: width-pair pool of xc (w even) / xtc (w odd)
  const int w0 = w & ~1;
  const float* sc = (w & 1) ? xtc : xc;
  const float a0 = sc[pc + h * W_ + w0];
  const float a1 = sc[pc + h * W_ + w0 + 1];
  const float ic = 0.5f * (a0 + a1) + fmaxf(a0, a1);
  // inter_row: height-pair pool of xc (h even) / xtc (h odd)
  const int h0 = h & ~1;
  const float* sr = (h & 1) ? xtc : xc;
  const float r0 = sr[pc + h0 * W_ + w];
  const float r1 = sr[pc + (h0 + 1) * W_ + w];
  const float ir = 0.5f * (r0 + r1) + fmaxf(r0, r1);

  const int lr = w * H_ + h;              // k0 index (W-major transpose)
  const int lr1 = w * H_ + (H_ - 1 - h);  // k1: flip of H inside ir
  const int lc = h * W_ + w;              // k2 index
  const int lc1 = h * W_ + (W_ - 1 - w);  // k3: flip of W inside inter_col

  const long zb = (long)b * KD;
  xs[((zb + 0) * DI + c) * L_ + lr] = ir;
  xs[((zb + 1) * DI + c) * L_ + lr1] = ir;
  xs[((zb + 2) * DI + c) * L_ + lc] = ic;
  xs[((zb + 3) * DI + c) * L_ + lc1] = ic;

  ut[((zb + 0) * L_ + lr) * DI + c] = ir;
  ut[((zb + 1) * L_ + lr1) * DI + c] = ir;
  ut[((zb + 2) * L_ + lc) * DI + c] = ic;
  ut[((zb + 3) * L_ + lc1) * DI + c] = ic;
}

// ------------------- delta = softplus(dts + bias), transpose ---------------
__global__ void delta_kernel(const float* __restrict__ dts,
                             const float* __restrict__ bias,
                             float* __restrict__ dtt) {
  const int idx = blockIdx.x * 256 + threadIdx.x;
  if (idx >= B_ * KD * L_ * DI) return;
  const int d = idx % DI;
  const int l = (idx / DI) % L_;
  const int z = idx / (DI * L_);
  const int kd = z & (KD - 1);
  const float x = dts[((long)z * DI + d) * L_ + l] + bias[kd * DI + d];
  const float sp = fmaxf(x, 0.f) + log1pf(__expf(-fabsf(x)));
  dtt[((long)z * L_ + l) * DI + d] = sp;
}

// ------------------- gather B/C rows of x_dbl into [z][l][16] --------------
__global__ void bc_kernel(const float* __restrict__ xdbl,
                          float* __restrict__ Bt, float* __restrict__ Ct) {
  const int idx = blockIdx.x * 256 + threadIdx.x;
  if (idx >= B_ * KD * L_ * DS_) return;
  const int n = idx % DS_;
  const int l = (idx / DS_) % L_;
  const int z = idx / (DS_ * L_);
  const long base = (long)z * C2 * L_;
  Bt[((long)z * L_ + l) * DS_ + n] = xdbl[base + (long)(DTR + n) * L_ + l];
  Ct[((long)z * L_ + l) * DS_ + n] = xdbl[base + (long)(DTR + DS_ + n) * L_ + l];
}

// ----------------------------- selective scan ------------------------------
// One lane per (b,k,d) channel, 16 states in registers. Sequential over L.
__global__ __launch_bounds__(DI) void scan_kernel(
    const float* __restrict__ ut, const float* __restrict__ dtt,
    const float* __restrict__ Bt, const float* __restrict__ Ct,
    const float* __restrict__ A_logs, const float* __restrict__ Ds,
    float* __restrict__ ys) {
  const int z = blockIdx.x;    // b*KD + kd
  const int d = threadIdx.x;   // channel
  const int kd = z & (KD - 1);
  const int row = kd * DI + d;
  float An[DS_], st[DS_];
#pragma unroll
  for (int n = 0; n < DS_; ++n) {
    An[n] = -__expf(A_logs[(long)row * DS_ + n]);
    st[n] = 0.f;
  }
  const float Dv = Ds[row];
  const long baseU = (long)z * L_ * DI;
  const long baseBC = (long)z * L_ * DS_;
  for (int l = 0; l < L_; ++l) {
    if (l + 8 < L_) {  // stream-ahead prefetch (global_prefetch_b8)
      __builtin_prefetch(ut + baseU + (long)(l + 8) * DI + d, 0, 1);
      __builtin_prefetch(dtt + baseU + (long)(l + 8) * DI + d, 0, 1);
    }
    const float u = ut[baseU + (long)l * DI + d];
    const float dl = dtt[baseU + (long)l * DI + d];
    const float4* Bp = (const float4*)(Bt + baseBC + (long)l * DS_);
    const float4* Cp = (const float4*)(Ct + baseBC + (long)l * DS_);
    float4 b4[4] = {Bp[0], Bp[1], Bp[2], Bp[3]};
    float4 c4[4] = {Cp[0], Cp[1], Cp[2], Cp[3]};
    const float* bb = (const float*)b4;
    const float* cc = (const float*)c4;
    float acc = 0.f;
#pragma unroll
    for (int n = 0; n < DS_; ++n) {
      const float dA = __expf(dl * An[n]);
      st[n] = st[n] * dA + dl * bb[n] * u;
      acc += st[n] * cc[n];
    }
    ys[baseU + (long)l * DI + d] = acc + Dv * u;
  }
}

// ----------------- directional merge + LayerNorm + SiLU gate ---------------
// ys layout [z][l][d]. One block per (b, l=h*W+w) position.
__global__ __launch_bounds__(128) void merge_ln_kernel(
    const float* __restrict__ ys, const float* __restrict__ xz,
    const float* __restrict__ g, const float* __restrict__ bt,
    float* __restrict__ y2) {
  const int m = blockIdx.x;  // b*L + l
  const int b = m / L_;
  const int l = m % L_;
  const int h = l / W_;
  const int w = l % W_;
  const int lr = w * H_ + h;
  __shared__ float red[128];
  const long base = (long)b * KD * L_ * DI;
  float vals[3];
  float sum = 0.f, sq = 0.f;
#pragma unroll
  for (int i = 0; i < 3; ++i) {
    const int d = threadIdx.x + i * 128;
    const float v = ys[base + ((long)0 * L_ + l) * DI + d] +
                    ys[base + ((long)2 * L_ + (L_ - 1 - l)) * DI + d] +
                    ys[base + ((long)1 * L_ + lr) * DI + d] +
                    ys[base + ((long)3 * L_ + (L_ - 1 - lr)) * DI + d];
    vals[i] = v;
    sum += v;
    sq += v * v;
  }
  red[threadIdx.x] = sum;
  __syncthreads();
  for (int s = 64; s > 0; s >>= 1) {
    if (threadIdx.x < s) red[threadIdx.x] += red[threadIdx.x + s];
    __syncthreads();
  }
  const float mu = red[0] * (1.f / DI);
  __syncthreads();
  red[threadIdx.x] = sq;
  __syncthreads();
  for (int s = 64; s > 0; s >>= 1) {
    if (threadIdx.x < s) red[threadIdx.x] += red[threadIdx.x + s];
    __syncthreads();
  }
  const float var = red[0] * (1.f / DI) - mu * mu;
  const float rinv = rsqrtf(var + 1e-5f);
#pragma unroll
  for (int i = 0; i < 3; ++i) {
    const int d = threadIdx.x + i * 128;
    const float zg = xz[(long)m * (2 * DI) + DI + d];
    const float silu = zg / (1.f + __expf(-zg));
    y2[(long)m * DI + d] = ((vals[i] - mu) * rinv * g[d] + bt[d]) * silu;
  }
}

// ---------------------------------------------------------------------------
extern "C" void kernel_launch(void* const* d_in, const int* in_sizes, int n_in,
                              void* d_out, int out_size, void* d_ws,
                              size_t ws_size, hipStream_t stream) {
  const float* x = (const float*)d_in[0];
  const float* xt = (const float*)d_in[1];
  const float* in_proj_w = (const float*)d_in[2];
  const float* conv2d_w = (const float*)d_in[3];
  const float* conv2d_b = (const float*)d_in[4];
  const float* conv2dxt_w = (const float*)d_in[5];
  const float* conv2dxt_b = (const float*)d_in[6];
  const float* x_proj_w = (const float*)d_in[7];
  const float* dt_projs_w = (const float*)d_in[8];
  const float* dt_projs_b = (const float*)d_in[9];
  const float* A_logs = (const float*)d_in[10];
  const float* Ds = (const float*)d_in[11];
  const float* out_norm_g = (const float*)d_in[12];
  const float* out_norm_b = (const float*)d_in[13];
  const float* out_proj_w = (const float*)d_in[14];

  float* ws = (float*)d_ws;
  size_t o = 0;
  float* xz = ws + o;   o += (size_t)M_TOT * 2 * DI;        // [9216][768]
  float* xc = ws + o;   o += (size_t)B_ * DI * L_;          // [B][DI][H][W]
  float* xtc = ws + o;  o += (size_t)B_ * DI * L_;
  float* xs = ws + o;   o += (size_t)B_ * KD * DI * L_;     // [z][d][l]
  float* xdbl = ws + o; o += (size_t)B_ * KD * C2 * L_;     // [z][44][l]
  float* dts = ws + o;  o += (size_t)B_ * KD * DI * L_;     // [z][d][l]
  float* ut = ws + o;   o += (size_t)B_ * KD * L_ * DI;     // [z][l][d]
  float* dtt = ws + o;  o += (size_t)B_ * KD * L_ * DI;     // [z][l][d]
  float* Btr = ws + o;  o += (size_t)B_ * KD * L_ * DS_;    // [z][l][16]
  float* Ctr = ws + o;  o += (size_t)B_ * KD * L_ * DS_;
  float* ys = xs;   // xs dead after x_dbl GEMM + interp
  float* y2 = dts;  // dts dead after delta_kernel

  // 1) in-proj GEMM: xz[9216,768] = x[9216,192] @ in_proj_w[768,192]^T
  wmma_gemm_kernel<true><<<dim3(768 / NT, M_TOT / MT, 1), 256, 0, stream>>>(
      x, in_proj_w, xz, M_TOT, 2 * DI, DM, DM, DM, 2 * DI, 0L, 1, 0L, 0L);

  // 2) convs (SiLU fused)
  const int totalC = B_ * H_ * W_ * DI;
  conv_dw_kernel<<<(totalC + 255) / 256, 256, 0, stream>>>(
      xz, 2 * DI, 0, conv2d_w, conv2d_b, xc);
  conv_dw_kernel<<<(totalC + 255) / 256, 256, 0, stream>>>(
      xt, DM, 1, conv2dxt_w, conv2dxt_b, xtc);

  // 3) directional interleave -> xs[z][d][l] and ut[z][l][d]
  interp_kernel<<<(totalC + 255) / 256, 256, 0, stream>>>(xc, xtc, xs, ut);

  // 4) x_dbl[z][44][l] = x_proj_w[kd][44][384] @ xs[z][384][l]  (16 batches)
  wmma_gemm_kernel<false><<<dim3(L_ / NT, 1, B_ * KD), 256, 0, stream>>>(
      x_proj_w, xs, xdbl, C2, L_, DI, DI, L_, L_, (long)C2 * DI, KD,
      (long)DI * L_, (long)C2 * L_);

  // 5) dts[z][384][l] = dt_projs_w[kd][384][12] @ x_dbl[z][0:12][l]
  wmma_gemm_kernel<false><<<dim3(L_ / NT, DI / MT, B_ * KD), 256, 0, stream>>>(
      dt_projs_w, xdbl, dts, DI, L_, DTR, DTR, L_, L_, (long)DI * DTR, KD,
      (long)C2 * L_, (long)DI * L_);

  // 6) delta = softplus(dts + bias), transposed to [z][l][d]
  const int totD = B_ * KD * L_ * DI;
  delta_kernel<<<(totD + 255) / 256, 256, 0, stream>>>(dts, dt_projs_b, dtt);

  // 7) gather B/C rows into [z][l][16]
  const int totBC = B_ * KD * L_ * DS_;
  bc_kernel<<<(totBC + 255) / 256, 256, 0, stream>>>(xdbl, Btr, Ctr);

  // 8) selective scan: 16 blocks x 384 lanes, 16 states/lane
  scan_kernel<<<B_ * KD, DI, 0, stream>>>(ut, dtt, Btr, Ctr, A_logs, Ds, ys);

  // 9) directional merge + LayerNorm + SiLU(z) gate -> y2[9216][384]
  merge_ln_kernel<<<M_TOT, 128, 0, stream>>>(ys, xz, out_norm_g, out_norm_b,
                                             y2);

  // 10) out-proj GEMM: out[9216,192] = y2[9216,384] @ out_proj_w[192,384]^T
  wmma_gemm_kernel<true><<<dim3(DM / NT, M_TOT / MT, 1), 256, 0, stream>>>(
      y2, out_proj_w, (float*)d_out, M_TOT, DM, DI, DI, DI, DM, 0L, 1, 0L, 0L);
}
